// RNN_90915867722100
// MI455X (gfx1250) — compile-verified, matrix operand order checked by reading, not run on hardware
//
#include <hip/hip_runtime.h>

#define T_STEPS 512
#define BATCH   256
#define HID     512
#define DIN     105
#define DPAD    128
#define NSENS   85
#define NRULE   20
#define NOUT    33
#define NOPAD   48
#define ALPHA   0.2f
#define SIGMA   0.05f

#define STEP_BLOCKS 128   // 512 waves: one per 16x16 tile of (B,H)
#define PROJ_BLOCKS 12    // 48 waves: one per 16x16 tile of (B,48)

typedef __attribute__((ext_vector_type(16))) __bf16         v16bf;
typedef __attribute__((ext_vector_type(8)))  float          v8f;
typedef __attribute__((ext_vector_type(8)))  unsigned short us8;
typedef __attribute__((ext_vector_type(16))) unsigned short us16;

union BF16x16 { us16 u; v16bf v; };

__device__ __forceinline__ unsigned short f2bf(float f) {
  unsigned int u = __float_as_uint(f);
  u += 0x7FFFu + ((u >> 16) & 1u);        // round-to-nearest-even
  return (unsigned short)(u >> 16);
}

// Build a 16-element bf16 A/B fragment from two contiguous 16-byte chunks.
// Matches the CDNA5 16-bit A (16x32) / B (32x16) lane layout:
//   lanes 0-15 take k-halves [ko, ko+8) and [ko+16, ko+24),
//   lanes 16-31 take [ko+8, ko+16) and [ko+24, ko+32) (caller adds hi*8).
__device__ __forceinline__ v16bf load_tile16(const unsigned short* p0,
                                             const unsigned short* p1) {
  us8 lo = *(const us8*)p0;
  us8 hi = *(const us8*)p1;
  BF16x16 t;
  #pragma unroll
  for (int i = 0; i < 8; ++i) { t.u[i] = lo[i]; t.u[8 + i] = hi[i]; }
  return t.v;
}

// ---- Preprocessing: pack x into bf16, zero-padded K=105 -> 128 -------------
__global__ void prep_x_kernel(const float* __restrict__ x,
                              unsigned short* __restrict__ xb) {
  const size_t N = (size_t)T_STEPS * BATCH * DPAD;
  for (size_t i = (size_t)blockIdx.x * blockDim.x + threadIdx.x; i < N;
       i += (size_t)gridDim.x * blockDim.x) {
    int    k  = (int)(i & (DPAD - 1));
    size_t tb = i >> 7;
    float  v  = (k < DIN) ? x[tb * DIN + k] : 0.0f;
    xb[i] = f2bf(v);
  }
}

// ---- Preprocessing: weights -> bf16 (Win = [Ws_w | Wr_w] padded,
//      Wrec*mask, Wout padded to 48 rows) ----------------------------------
__global__ void prep_w_kernel(const float* __restrict__ Ws_w,
                              const float* __restrict__ Wr_w,
                              const float* __restrict__ Wrec_w,
                              const float* __restrict__ mask,
                              const float* __restrict__ Wout_w,
                              unsigned short* __restrict__ win,
                              unsigned short* __restrict__ wrec,
                              unsigned short* __restrict__ wout) {
  const int NW = HID * DPAD;      // 65536
  const int NR = HID * HID;       // 262144
  const int NQ = NOPAD * HID;     // 24576
  const int total = NW + NR + NQ;
  for (int i = blockIdx.x * blockDim.x + threadIdx.x; i < total;
       i += gridDim.x * blockDim.x) {
    if (i < NW) {
      int h = i / DPAD, k = i - h * DPAD;
      float v = (k < NSENS) ? Ws_w[h * NSENS + k]
              : (k < DIN)   ? Wr_w[h * NRULE + (k - NSENS)]
                            : 0.0f;
      win[i] = f2bf(v);
    } else if (i < NW + NR) {
      int j = i - NW;
      wrec[j] = f2bf(Wrec_w[j] * mask[j]);
    } else {
      int j = i - NW - NR;
      int n = j / HID;
      wout[j] = f2bf((n < NOUT) ? Wout_w[j] : 0.0f);
    }
  }
}

__global__ void init_h_kernel(float* __restrict__ hf,
                              unsigned short* __restrict__ hb) {
  int i = blockIdx.x * blockDim.x + threadIdx.x;
  if (i < BATCH * HID) { hf[i] = 0.0f; hb[i] = 0; }
}

// ---- Output projection tile: out[t, 16x16 tile] = h @ Wout^T + b -----------
// wave in [0,48): tileB = wave/3 (B rows), tileN = wave%3 (48 padded cols).
__device__ __forceinline__ void proj_tile(int t, int wave, int lane,
                                          const unsigned short* __restrict__ hb,
                                          const unsigned short* __restrict__ wout,
                                          const float* __restrict__ Wout_b,
                                          float* __restrict__ out) {
  int tileB = wave / 3;
  int tileN = wave % 3;
  int bm = tileB * 16, nn = tileN * 16;
  int lr = lane & 15, hi = lane >> 4;

  v8f acc = {};
  const unsigned short* arow = hb   + (size_t)(bm + lr) * HID;
  const unsigned short* brow = wout + (size_t)(nn + lr) * HID;
  #pragma unroll
  for (int kt = 0; kt < HID / 32; ++kt) {
    int ko = kt * 32 + hi * 8;
    v16bf a = load_tile16(arow + ko, arow + ko + 16);
    v16bf b = load_tile16(brow + ko, brow + ko + 16);
    acc = __builtin_amdgcn_wmma_f32_16x16x32_bf16(false, a, false, b,
                                                  (short)0, acc, false, false);
  }

  int n = nn + lr;
  if (n < NOUT) {
    float bias = Wout_b[n];
    #pragma unroll
    for (int r = 0; r < 8; ++r) {
      int m = bm + hi * 8 + r;
      out[((size_t)t * BATCH + m) * NOUT + n] = acc[r] + bias;
    }
  }
}

// ---- Fused step: blocks [0,128) compute h_new for step t; blocks [128,140)
//      concurrently project out[t-1] from h_old (already complete). ---------
__global__ __launch_bounds__(128)
void rnn_step_fused_kernel(int t,
                           const unsigned short* __restrict__ xb,
                           const unsigned short* __restrict__ win,
                           const unsigned short* __restrict__ wrec,
                           const float* __restrict__ noise,
                           const float* __restrict__ Ws_b,
                           const float* __restrict__ Wrec_b,
                           const float* __restrict__ hf_old,
                           const unsigned short* __restrict__ hb_old,
                           float* __restrict__ hf_new,
                           unsigned short* __restrict__ hb_new,
                           const unsigned short* __restrict__ wout,
                           const float* __restrict__ Wout_b,
                           float* __restrict__ out) {
  int lane = threadIdx.x & 31;

  if (blockIdx.x >= STEP_BLOCKS) {
    // Projection of the previous step's hidden state (independent work).
    if (t == 0) return;
    int wave = ((blockIdx.x - STEP_BLOCKS) * blockDim.x + threadIdx.x) >> 5;
    proj_tile(t - 1, wave, lane, hb_old, wout, Wout_b, out);
    return;
  }

  int wave  = (blockIdx.x * blockDim.x + threadIdx.x) >> 5;
  int tileB = wave >> 5;            // 0..15
  int tileH = wave & 31;            // 0..31
  int bm = tileB * 16, hn = tileH * 16;
  int lr = lane & 15, hi = lane >> 4;

  v8f acc = {};

  // Input projection: A = x_t (16x128 bf16), B = Win^T (rows of Win)
  const unsigned short* xrow = xb + ((size_t)t * BATCH + (bm + lr)) * DPAD;
  const unsigned short* wrow = win + (size_t)(hn + lr) * DPAD;
  #pragma unroll
  for (int kt = 0; kt < DPAD / 32; ++kt) {
    int ko = kt * 32 + hi * 8;
    v16bf a = load_tile16(xrow + ko, xrow + ko + 16);
    v16bf b = load_tile16(wrow + ko, wrow + ko + 16);
    acc = __builtin_amdgcn_wmma_f32_16x16x32_bf16(false, a, false, b,
                                                  (short)0, acc, false, false);
  }

  // Recurrent projection: A = h_prev (16x512 bf16), B = Wrec^T (rows of Wrec)
  const unsigned short* hrow = hb_old + (size_t)(bm + lr) * HID;
  const unsigned short* rrow = wrec   + (size_t)(hn + lr) * HID;
  #pragma unroll
  for (int kt = 0; kt < HID / 32; ++kt) {
    int ko = kt * 32 + hi * 8;
    v16bf a = load_tile16(hrow + ko, hrow + ko + 16);
    v16bf b = load_tile16(rrow + ko, rrow + ko + 16);
    acc = __builtin_amdgcn_wmma_f32_16x16x32_bf16(false, a, false, b,
                                                  (short)0, acc, false, false);
  }

  // Epilogue: bias + sigma*noise + relu + leaky blend. C/D layout:
  // VGPR r holds (M = r + 8*hi, N = lane&15) of the 16x16 tile.
  int   n    = hn + lr;
  float bias = Ws_b[n] + Wrec_b[n];
  #pragma unroll
  for (int r = 0; r < 8; ++r) {
    int    m    = bm + hi * 8 + r;
    size_t hidx = (size_t)m * HID + n;
    float pre  = acc[r] + bias + noise[((size_t)t * BATCH + m) * HID + n] * SIGMA;
    float outv = fmaxf(pre, 0.0f);
    float hnew = ALPHA * outv + (1.0f - ALPHA) * hf_old[hidx];
    hf_new[hidx] = hnew;
    hb_new[hidx] = f2bf(hnew);
  }
}

// ---- Standalone projection (used once for the final timestep) --------------
__global__ __launch_bounds__(128)
void out_proj_kernel(int t,
                     const unsigned short* __restrict__ hb,
                     const unsigned short* __restrict__ wout,
                     const float* __restrict__ Wout_b,
                     float* __restrict__ out) {
  int wave = (blockIdx.x * blockDim.x + threadIdx.x) >> 5;
  int lane = threadIdx.x & 31;
  proj_tile(t, wave, lane, hb, wout, Wout_b, out);
}

extern "C" void kernel_launch(void* const* d_in, const int* in_sizes, int n_in,
                              void* d_out, int out_size, void* d_ws, size_t ws_size,
                              hipStream_t stream) {
  const float* x      = (const float*)d_in[0];
  const float* noise  = (const float*)d_in[1];
  const float* Ws_w   = (const float*)d_in[2];
  const float* Ws_b   = (const float*)d_in[3];
  const float* Wr_w   = (const float*)d_in[4];
  const float* Wrec_w = (const float*)d_in[5];
  const float* Wrec_b = (const float*)d_in[6];
  const float* mask   = (const float*)d_in[7];
  const float* Wout_w = (const float*)d_in[8];
  const float* Wout_b = (const float*)d_in[9];
  float* out = (float*)d_out;

  // Workspace layout (~35.8 MB total)
  char* p = (char*)d_ws;
  auto alloc = [&](size_t bytes) {
    char* r = p;
    p += (bytes + 255) & ~(size_t)255;
    return r;
  };
  unsigned short* xb   = (unsigned short*)alloc((size_t)T_STEPS * BATCH * DPAD * 2);
  unsigned short* win  = (unsigned short*)alloc((size_t)HID * DPAD * 2);
  unsigned short* wrec = (unsigned short*)alloc((size_t)HID * HID * 2);
  unsigned short* wout = (unsigned short*)alloc((size_t)NOPAD * HID * 2);
  float*          hf0  = (float*)alloc((size_t)BATCH * HID * 4);
  float*          hf1  = (float*)alloc((size_t)BATCH * HID * 4);
  unsigned short* hb0  = (unsigned short*)alloc((size_t)BATCH * HID * 2);
  unsigned short* hb1  = (unsigned short*)alloc((size_t)BATCH * HID * 2);
  float*          hf[2] = { hf0, hf1 };
  unsigned short* hb[2] = { hb0, hb1 };

  prep_x_kernel<<<2048, 256, 0, stream>>>(x, xb);
  prep_w_kernel<<<512, 256, 0, stream>>>(Ws_w, Wr_w, Wrec_w, mask, Wout_w,
                                         win, wrec, wout);
  init_h_kernel<<<512, 256, 0, stream>>>(hf[0], hb[0]);

  for (int t = 0; t < T_STEPS; ++t) {
    int cur = t & 1, nxt = cur ^ 1;
    // Blocks [0,128): step t. Blocks [128,140): out[t-1] from h[cur].
    rnn_step_fused_kernel<<<STEP_BLOCKS + PROJ_BLOCKS, 128, 0, stream>>>(
        t, xb, win, wrec, noise, Ws_b, Wrec_b,
        hf[cur], hb[cur], hf[nxt], hb[nxt],
        wout, Wout_b, out);
  }
  // Final projection: out[511] from h after the last step (buffer parity:
  // step t=511 has cur=1, writes nxt=0).
  out_proj_kernel<<<PROJ_BLOCKS, 128, 0, stream>>>(T_STEPS - 1, hb[0],
                                                   wout, Wout_b, out);
}
